// GNN_7_44684839747778
// MI455X (gfx1250) — compile-verified, matrix operand order checked by reading, not run on hardware
//
#include <hip/hip_runtime.h>
#include <hip/hip_bf16.h>
#include <stdint.h>

#define NN 10000
#define NE 160000
#define NG 100

typedef __attribute__((ext_vector_type(16))) __bf16 v16bf;
typedef __attribute__((ext_vector_type(8)))  float  v8f;

union FragBF { uint4 q[2]; v16bf v; };

static __device__ __forceinline__ unsigned short f2bf(float f) {
    union { float f; unsigned int u; } v; v.f = f;
    unsigned int r = v.u + 0x7FFFu + ((v.u >> 16) & 1u);   // round-to-nearest-even
    return (unsigned short)(r >> 16);
}

__global__ void k_zero(float* __restrict__ p, int n) {
    int i = blockIdx.x * blockDim.x + threadIdx.x;
    if (i < n) p[i] = 0.0f;
}

// agg[dst] += x[src] * edge_attr ; one thread per (edge, 4-channel chunk)
__global__ void k_edge_scatter(const float* __restrict__ x, const int* __restrict__ src,
                               const int* __restrict__ dst, const float* __restrict__ ea,
                               float* __restrict__ agg, int cin, int total) {
    int i = blockIdx.x * blockDim.x + threadIdx.x;
    if (i >= total) return;
    int chunks = cin >> 2;
    int e = i / chunks;
    int c = i - e * chunks;
    int s = src[e], d = dst[e];
    float w = ea[e];
    const float4 v = *reinterpret_cast<const float4*>(x + (size_t)s * cin + c * 4);
    float* o = agg + (size_t)d * cin + c * 4;
    atomicAdd(o + 0, v.x * w);
    atomicAdd(o + 1, v.y * w);
    atomicAdd(o + 2, v.z * w);
    atomicAdd(o + 3, v.w * w);
}

// fp32 -> bf16 for both x and agg, zero-padded to kpad columns
__global__ void k_cvt2bf(const float* __restrict__ X, const float* __restrict__ A,
                         unsigned short* __restrict__ Xb, unsigned short* __restrict__ Ab,
                         int cin, int kpad, int total) {
    int i = blockIdx.x * blockDim.x + threadIdx.x;
    if (i >= total) return;
    int row = i / kpad;
    int k = i - row * kpad;
    unsigned short xv = 0, av = 0;
    if (k < cin) {
        size_t o = (size_t)row * cin + k;
        xv = f2bf(X[o]);
        av = f2bf(A[o]);
    }
    size_t q = (size_t)row * kpad + k;
    Xb[q] = xv;
    Ab[q] = av;
}

// W[cin][cout] fp32 -> W^T[cout][kpad] bf16 (both Wr and Wo), zero-padded K
__global__ void k_cvtW(const float* __restrict__ Wr, const float* __restrict__ Wo,
                       unsigned short* __restrict__ WrT, unsigned short* __restrict__ WoT,
                       int cin, int cout, int kpad, int total) {
    int i = blockIdx.x * blockDim.x + threadIdx.x;
    if (i >= total) return;
    int co = i / kpad;
    int k = i - co * kpad;
    unsigned short r = 0, o = 0;
    if (k < cin) {
        size_t off = (size_t)k * cout + co;
        r = f2bf(Wr[off]);
        o = f2bf(Wo[off]);
    }
    size_t q = (size_t)co * kpad + k;
    WrT[q] = r;
    WoT[q] = o;
}

// out = relu(agg @ Wr + x @ Wo + b)
// One wave per 16 x (16*NT) output tile: A fragments (agg row, x row) are loaded
// once per K-step and reused across NT column sub-tiles -> 2*NT WMMAs per step.
template <int NT>
__global__ void __launch_bounds__(32)
k_wmma_dualgemm(const unsigned short* __restrict__ Ab,   // agg bf16 [NN][kpad]
                const unsigned short* __restrict__ Xb,   // x   bf16 [NN][kpad]
                const unsigned short* __restrict__ WrT,  // Wr^T bf16 [cout][kpad]
                const unsigned short* __restrict__ WoT,  // Wo^T bf16 [cout][kpad]
                const float* __restrict__ bias,
                float* __restrict__ out, int kpad, int cout) {
    const int lane = threadIdx.x & 31;
    const int l15  = lane & 15;
    const int half = lane >> 4;                 // wave32: two half-wave K groups
    const int row  = blockIdx.x * 16 + l15;     // A fragment: M = lane&15
    const int n0   = blockIdx.y * 16 * NT;      // first output column of this wave
    const size_t arow = (size_t)row * kpad;

    size_t brow[NT];
#pragma unroll
    for (int t = 0; t < NT; t++) brow[t] = (size_t)(n0 + t * 16 + l15) * kpad;

    v8f acc[NT];
#pragma unroll
    for (int t = 0; t < NT; t++) acc[t] = (v8f){};

    for (int k = 0; k < kpad; k += 32) {
        // 16-bit A-matrix 16x32 layout: VGPR0-3 <- K = half*8+0..7, VGPR4-7 <- K = 16+half*8+0..7
        const int k0 = k + half * 8;
        const int k1 = k + 16 + half * 8;

        if (k + 32 < kpad) {   // software prefetch next K-block of the A streams
            __builtin_prefetch(Ab + arow + k0 + 32, 0, 1);
            __builtin_prefetch(Xb + arow + k0 + 32, 0, 1);
        }

        FragBF aA, aX;
        aA.q[0] = *reinterpret_cast<const uint4*>(Ab + arow + k0);
        aA.q[1] = *reinterpret_cast<const uint4*>(Ab + arow + k1);
        aX.q[0] = *reinterpret_cast<const uint4*>(Xb + arow + k0);
        aX.q[1] = *reinterpret_cast<const uint4*>(Xb + arow + k1);

#pragma unroll
        for (int t = 0; t < NT; t++) {
            FragBF bR, bO;
            bR.q[0] = *reinterpret_cast<const uint4*>(WrT + brow[t] + k0);
            bR.q[1] = *reinterpret_cast<const uint4*>(WrT + brow[t] + k1);
            bO.q[0] = *reinterpret_cast<const uint4*>(WoT + brow[t] + k0);
            bO.q[1] = *reinterpret_cast<const uint4*>(WoT + brow[t] + k1);
            acc[t] = __builtin_amdgcn_wmma_f32_16x16x32_bf16(false, aA.v, false, bR.v,
                                                             (short)0, acc[t], false, false);
            acc[t] = __builtin_amdgcn_wmma_f32_16x16x32_bf16(false, aX.v, false, bO.v,
                                                             (short)0, acc[t], false, false);
        }
    }

    // C/D layout: lanes 0-15 -> M = 0..7 (VGPR r), lanes 16-31 -> M = 8..15
    const int mb = half * 8;
#pragma unroll
    for (int t = 0; t < NT; t++) {
        const int n  = n0 + t * 16 + l15;
        const float bb = bias[n];
#pragma unroll
        for (int r = 0; r < 8; r++) {
            float v = acc[t][r] + bb;
            v = v > 0.0f ? v : 0.0f;
            out[(size_t)(blockIdx.x * 16 + mb + r) * cout + n] = v;
        }
    }
}

__global__ void k_count(const int* __restrict__ batch, float* __restrict__ counts) {
    int i = blockIdx.x * blockDim.x + threadIdx.x;
    if (i < NN) atomicAdd(&counts[batch[i]], 1.0f);
}

__global__ void k_poolsum(const float* __restrict__ X, const int* __restrict__ batch,
                          float* __restrict__ pooled, int c, int total) {
    int i = blockIdx.x * blockDim.x + threadIdx.x;
    if (i >= total) return;
    int chunks = c >> 2;
    int node = i / chunks;
    int q = i - node * chunks;
    int g = batch[node];
    const float4 v = *reinterpret_cast<const float4*>(X + (size_t)node * c + q * 4);
    float* o = pooled + (size_t)g * c + q * 4;
    atomicAdd(o + 0, v.x);
    atomicAdd(o + 1, v.y);
    atomicAdd(o + 2, v.z);
    atomicAdd(o + 3, v.w);
}

__global__ void k_pooldiv(const float* __restrict__ pooled, const float* __restrict__ counts,
                          float* __restrict__ h, int c, int total) {
    int i = blockIdx.x * blockDim.x + threadIdx.x;
    if (i >= total) return;
    int g = i / c;
    float cnt = counts[g];
    cnt = cnt > 1.0f ? cnt : 1.0f;
    h[i] = pooled[i] / cnt;
}

__global__ void k_mlp(const float* __restrict__ in, const float* __restrict__ W,
                      const float* __restrict__ b, float* __restrict__ out,
                      int cin, int cout, int dorelu, int total) {
    int i = blockIdx.x * blockDim.x + threadIdx.x;
    if (i >= total) return;
    int g = i / cout;
    int co = i - g * cout;
    float s = b[co];
    const float* row = in + (size_t)g * cin;
    for (int k = 0; k < cin; k++) s += row[k] * W[(size_t)k * cout + co];
    if (dorelu) s = s > 0.0f ? s : 0.0f;
    out[i] = s;
}

extern "C" void kernel_launch(void* const* d_in, const int* in_sizes, int n_in,
                              void* d_out, int out_size, void* d_ws, size_t ws_size,
                              hipStream_t stream) {
    static const int DIMS[8] = {4, 32, 128, 256, 512, 512, 256, 256};

    const float* x0    = (const float*)d_in[0];
    const int*   src   = (const int*)d_in[1];          // edge_index row 0
    const int*   dst   = src + NE;                     // edge_index row 1
    const float* ea    = (const float*)d_in[2];
    const int*   batch = (const int*)d_in[3];
    const float* Wr[7]; const float* Wo[7]; const float* Bb[7];
    for (int i = 0; i < 7; i++) {
        Wr[i] = (const float*)d_in[4 + 3 * i];
        Wo[i] = (const float*)d_in[5 + 3 * i];
        Bb[i] = (const float*)d_in[6 + 3 * i];
    }
    const float* LW[4]; const float* LB[4];
    for (int i = 0; i < 4; i++) {
        LW[i] = (const float*)d_in[25 + 2 * i];
        LB[i] = (const float*)d_in[26 + 2 * i];
    }

    // ---- workspace carve (max feature dim 512) ----
    char* ws = (char*)d_ws;
    size_t off = 0;
    auto carve = [&](size_t bytes) -> char* {
        char* p = ws + off;
        off += (bytes + 255) & ~(size_t)255;
        return p;
    };
    float*          bufA   = (float*)carve((size_t)NN * 512 * 4);
    float*          bufB   = (float*)carve((size_t)NN * 512 * 4);
    float*          agg    = (float*)carve((size_t)NN * 512 * 4);
    unsigned short* Xb     = (unsigned short*)carve((size_t)NN * 512 * 2);
    unsigned short* Abf    = (unsigned short*)carve((size_t)NN * 512 * 2);
    unsigned short* WrT    = (unsigned short*)carve((size_t)512 * 512 * 2);
    unsigned short* WoT    = (unsigned short*)carve((size_t)512 * 512 * 2);
    float*          pooled = (float*)carve((size_t)NG * 256 * 4);
    float*          counts = (float*)carve((size_t)NG * 4);
    float*          m1     = (float*)carve((size_t)NG * 256 * 4);
    float*          m2     = (float*)carve((size_t)NG * 256 * 4);

    const int TB = 256;
    const float* cur = x0;
    float* nxt = bufA;

    for (int l = 0; l < 7; l++) {
        int cin  = DIMS[l];
        int cout = DIMS[l + 1];
        int kpad = ((cin + 31) / 32) * 32;

        int nz = NN * cin;
        k_zero<<<(nz + TB - 1) / TB, TB, 0, stream>>>(agg, nz);

        int te = NE * (cin / 4);
        k_edge_scatter<<<(te + TB - 1) / TB, TB, 0, stream>>>(cur, src, dst, ea, agg, cin, te);

        int tc = NN * kpad;
        k_cvt2bf<<<(tc + TB - 1) / TB, TB, 0, stream>>>(cur, agg, Xb, Abf, cin, kpad, tc);

        int tw = cout * kpad;
        k_cvtW<<<(tw + TB - 1) / TB, TB, 0, stream>>>(Wr[l], Wo[l], WrT, WoT, cin, cout, kpad, tw);

        if (cout % 64 == 0) {
            dim3 grid(NN / 16, cout / 64);   // 10000 = 625*16 exactly
            k_wmma_dualgemm<4><<<grid, 32, 0, stream>>>(Abf, Xb, WrT, WoT, Bb[l], nxt, kpad, cout);
        } else {
            dim3 grid(NN / 16, cout / 32);   // layer 1: cout = 32
            k_wmma_dualgemm<2><<<grid, 32, 0, stream>>>(Abf, Xb, WrT, WoT, Bb[l], nxt, kpad, cout);
        }

        cur = nxt;
        nxt = (nxt == bufA) ? bufB : bufA;
    }

    // ---- global mean pool over graphs (cur is [NN, 256]) ----
    k_zero<<<(NG * 256 + TB - 1) / TB, TB, 0, stream>>>(pooled, NG * 256);
    k_zero<<<1, TB, 0, stream>>>(counts, NG);
    k_count<<<(NN + TB - 1) / TB, TB, 0, stream>>>(batch, counts);
    int tp = NN * (256 / 4);
    k_poolsum<<<(tp + TB - 1) / TB, TB, 0, stream>>>(cur, batch, pooled, 256, tp);
    k_pooldiv<<<(NG * 256 + TB - 1) / TB, TB, 0, stream>>>(pooled, counts, m1, 256, NG * 256);

    // ---- MLP head 256 -> 256 -> 128 -> 64 -> 1 ----
    k_mlp<<<(NG * 256 + TB - 1) / TB, TB, 0, stream>>>(m1, LW[0], LB[0], m2, 256, 256, 1, NG * 256);
    k_mlp<<<(NG * 128 + TB - 1) / TB, TB, 0, stream>>>(m2, LW[1], LB[1], m1, 256, 128, 1, NG * 128);
    k_mlp<<<(NG * 64  + TB - 1) / TB, TB, 0, stream>>>(m1, LW[2], LB[2], m2, 128, 64, 1, NG * 64);
    k_mlp<<<1, TB, 0, stream>>>(m2, LW[3], LB[3], (float*)d_out, 64, 1, 0, NG);
}
